// Quantizer_DPgQ_40243843563650
// MI455X (gfx1250) — matrix operand order repaired and checked by reading, not compiled, not run on hardware
//
#include <hip/hip_runtime.h>
#include <cstdint>

typedef _Float16 v16h __attribute__((ext_vector_type(16)));
typedef float    v8f  __attribute__((ext_vector_type(8)));
typedef unsigned int v4u __attribute__((ext_vector_type(4)));
typedef int      v4i  __attribute__((ext_vector_type(4)));
typedef int      v8i  __attribute__((ext_vector_type(8)));

#define N_ROWS  32768
#define DDIM    512
#define QSTAGES 8
#define KCODES  4096
#define TILE_M  128
#define THREADS 256            // 8 waves, each owns a 16-row strip
#define NBLK    (KCODES / 16)  // 256 column blocks per stage

// Padded LDS layout for the TDM-staged B tile: 16 rows x 1024B, +32B pad per
// row (TDM pad_interval=7 -> every 256 DWORDs, pad_amount=7 -> 8 DWORDs).
// Row stride 1056B keeps v16h fragments 32B-aligned and spreads LDS banks.
#define BROW_STRIDE 1056
#define BBUF_BYTES  (16 * BROW_STRIDE)
#define NBUF        3

// ---------------------------------------------------------------------------
// Kernel 1: L2-normalize every codeword row into f16. One wave32 per row.
// ---------------------------------------------------------------------------
__global__ __launch_bounds__(256) void cb_norm_kernel(
    const float* __restrict__ cb, _Float16* __restrict__ cb16)
{
    const int gwave = (int)((blockIdx.x * blockDim.x + threadIdx.x) >> 5);
    const int lane  = (int)(threadIdx.x & 31);
    if (gwave >= QSTAGES * KCODES) return;

    const float* row = cb + (size_t)gwave * DDIM;
    const float4* r4 = (const float4*)(row + lane * 16);
    float4 f0 = r4[0], f1 = r4[1], f2 = r4[2], f3 = r4[3];
    float s = f0.x*f0.x + f0.y*f0.y + f0.z*f0.z + f0.w*f0.w
            + f1.x*f1.x + f1.y*f1.y + f1.z*f1.z + f1.w*f1.w
            + f2.x*f2.x + f2.y*f2.y + f2.z*f2.z + f2.w*f2.w
            + f3.x*f3.x + f3.y*f3.y + f3.z*f3.z + f3.w*f3.w;
    #pragma unroll
    for (int m = 16; m >= 1; m >>= 1) s += __shfl_xor(s, m, 32);
    const float inv = 1.0f / fmaxf(sqrtf(s), 1e-12f);

    v16h h;
    h[0]=(_Float16)(f0.x*inv);  h[1]=(_Float16)(f0.y*inv);
    h[2]=(_Float16)(f0.z*inv);  h[3]=(_Float16)(f0.w*inv);
    h[4]=(_Float16)(f1.x*inv);  h[5]=(_Float16)(f1.y*inv);
    h[6]=(_Float16)(f1.z*inv);  h[7]=(_Float16)(f1.w*inv);
    h[8]=(_Float16)(f2.x*inv);  h[9]=(_Float16)(f2.y*inv);
    h[10]=(_Float16)(f2.z*inv); h[11]=(_Float16)(f2.w*inv);
    h[12]=(_Float16)(f3.x*inv); h[13]=(_Float16)(f3.y*inv);
    h[14]=(_Float16)(f3.z*inv); h[15]=(_Float16)(f3.w*inv);
    *(v16h*)(cb16 + (size_t)gwave * DDIM + lane * 16) = h;
}

// ---------------------------------------------------------------------------
// TDM issue: 1-D tile of 8192 f16 elements (one 16-column codebook block,
// 16 KB contiguous) DMA'd global -> LDS with row padding. D# per ISA §8.
// ---------------------------------------------------------------------------
__device__ __forceinline__ void tdm_issue(uint64_t gaddr, uint32_t laddr,
                                          v8i g1) {
    v4u g0;
    g0.x = 1u;                                   // count=1, user descriptor
    g0.y = laddr;                                // lds_addr (bytes)
    g0.z = (uint32_t)gaddr;                      // global_addr[31:0]
    g0.w = (uint32_t)((gaddr >> 32) & 0x01FFFFFFu) | (2u << 30); // addr[56:32], type=2
    v4i z4 = {0, 0, 0, 0};
#if defined(__clang_major__) && __clang_major__ >= 23
    v8i z8 = {0, 0, 0, 0, 0, 0, 0, 0};
    __builtin_amdgcn_tensor_load_to_lds(g0, g1, z4, z4, z8, 0);
#else
    __builtin_amdgcn_tensor_load_to_lds(g0, g1, z4, z4, 0);
#endif
}

// ---------------------------------------------------------------------------
// Kernel 2: residual VQ over a 128-row tile, all 8 stages.
// Dynamic LDS: fp32 residual [128][512] (256 KB) + 3 TDM B buffers (50.7 KB).
// ---------------------------------------------------------------------------
#define CVT8(dst, off, A, B)                                                \
    dst[(off)+0]=(_Float16)((A).x*inv); dst[(off)+1]=(_Float16)((A).y*inv); \
    dst[(off)+2]=(_Float16)((A).z*inv); dst[(off)+3]=(_Float16)((A).w*inv); \
    dst[(off)+4]=(_Float16)((B).x*inv); dst[(off)+5]=(_Float16)((B).y*inv); \
    dst[(off)+6]=(_Float16)((B).z*inv); dst[(off)+7]=(_Float16)((B).w*inv);

__global__ __launch_bounds__(THREADS) void rvq_kernel(
    const float* __restrict__ x, const float* __restrict__ cb,
    const _Float16* __restrict__ cb16,
    float* __restrict__ outHard, float* __restrict__ outIds)
{
    extern __shared__ float smem[];
    float* res = smem;                               // [TILE_M][DDIM] fp32
    char*  bptr0 = (char*)(smem + TILE_M * DDIM);    // 3 contiguous B buffers
    __shared__ int bestIdxLds[TILE_M];

    const uint32_t bOff0 = (uint32_t)(uintptr_t)bptr0;   // LDS byte offset

    const int tid  = (int)threadIdx.x;
    const int lane = tid & 31;
    const int wave = tid >> 5;
    const int row0 = (int)blockIdx.x * TILE_M;

    // Loop-invariant D# group 1: data_size=2B, pad every 256 DW by 8 DW,
    // tensor/tile dim0 = 8192 elems, dim1 = 1 row, stride 8192.
    v8i g1;
    g1[0] = (1 << 16) | (1 << 20) | (7 << 22) | (7 << 25);
    g1[1] = (int)(8192u << 16);      // tensor_dim0[15:0] in bits 63:48
    g1[2] = (int)(1u << 16);         // tensor_dim0 hi=0; tensor_dim1=1
    g1[3] = (int)(8192u << 16);      // tensor_dim1 hi=0; tile_dim0=8192
    g1[4] = 1;                       // tile_dim1=1, tile_dim2=0
    g1[5] = 8192;                    // tensor_dim0_stride[31:0]
    g1[6] = (int)(8192u << 16);      // stride0 hi=0; tensor_dim1_stride lo
    g1[7] = 0;                       // tensor_dim1_stride hi

    // residual := x tile
    {
        const float4* xs = (const float4*)(x + (size_t)row0 * DDIM);
        float4* rd = (float4*)res;
        for (int i = tid; i < TILE_M * DDIM / 4; i += THREADS) rd[i] = xs[i];
    }
    __syncthreads();

    const int rowL    = lane & 15;              // A row / B column within strip
    const int koff    = (lane < 16) ? 0 : 16;   // A K sub-offset (contiguous k = 16h + s)
    const int koff2   = (lane < 16) ? 0 : 16;   // B K sub-offset (contiguous k = 16h + s)
    const int waveRow = wave * 16;
    const int laneB   = rowL * BROW_STRIDE + koff2 * 2;  // lane's B-frag base

    for (int q = 0; q < QSTAGES; q++) {
        const _Float16* cbq = cb16 + (size_t)q * KCODES * DDIM;
        const uint64_t cbqAddr = (uint64_t)(uintptr_t)cbq;

        // Prefetch first B block via TDM; overlaps with the norm pass below.
        // (Buffer 0 was last read at iter NBLK-1 of the previous stage, which
        //  is sealed by that stage's trailing __syncthreads.)
        if (wave == 0) tdm_issue(cbqAddr, bOff0, g1);

        const float* rrow = res + (size_t)(waveRow + rowL) * DDIM;

        // ---- pass 1: sum of squares over this lane's half of the row ----
        float ssum = 0.f;
        #pragma unroll
        for (int kc = 0; kc < 16; kc++) {
            const float4* p = (const float4*)(rrow + kc * 32 + koff);
            float4 a0 = p[0], a1 = p[1], a2 = p[2], a3 = p[3];
            ssum += a0.x*a0.x + a0.y*a0.y + a0.z*a0.z + a0.w*a0.w
                  + a1.x*a1.x + a1.y*a1.y + a1.z*a1.z + a1.w*a1.w
                  + a2.x*a2.x + a2.y*a2.y + a2.z*a2.z + a2.w*a2.w
                  + a3.x*a3.x + a3.y*a3.y + a3.z*a3.z + a3.w*a3.w;
        }
        ssum += __shfl_xor(ssum, 16, 32);       // lane L + L+16 = full row
        const float inv = 1.0f / fmaxf(sqrtf(ssum), 1e-12f);

        // ---- pass 2: cache all 16 normalized-f16 A fragments in VGPRs ----
        v16h afr[16];
        #pragma unroll
        for (int kc = 0; kc < 16; kc++) {
            const float4* p = (const float4*)(rrow + kc * 32 + koff);
            float4 a0 = p[0], a1 = p[1], a2 = p[2], a3 = p[3];
            CVT8(afr[kc], 0, a0, a1)
            CVT8(afr[kc], 8, a2, a3)
        }

        // ---- sweep 4096 codewords: triple-buffered TDM + pipelined WMMA ----
        float bestv[8];
        int   besti[8];
        #pragma unroll
        for (int r = 0; r < 8; r++) { bestv[r] = -1e30f; besti[r] = 0; }

        uint32_t roff = 0;                       // buffer offset being read
        uint32_t woff = BBUF_BYTES;              // buffer offset being written

        for (int nb = 0; nb < NBLK; nb++) {
            if (wave == 0) {
                if (nb + 1 < NBLK) {
                    tdm_issue(cbqAddr + (uint64_t)(nb + 1) * (16 * DDIM * 2),
                              bOff0 + woff, g1);
                    __builtin_amdgcn_s_wait_tensorcnt((short)1);
                } else {
                    __builtin_amdgcn_s_wait_tensorcnt((short)0);
                }
            }
            __syncthreads();   // buffer at roff is resident for all waves;
                               // (3 buffers: the woff target was last read two
                               //  iterations ago, sealed by this barrier chain)

            const char* bbase = bptr0 + roff + laneB;
            const int col = nb * 16 + rowL;
            v8f acc = {};
            // depth-2 software pipeline: keep 2 B fragments in flight per WMMA
            v16h b0 = *(const v16h*)(bbase);
            v16h b1 = *(const v16h*)(bbase + 64);
            #pragma unroll
            for (int kc = 0; kc < 16; kc++) {
                v16h b2 = b1;
                if (kc < 14) b2 = *(const v16h*)(bbase + (kc + 2) * 64);
                acc = __builtin_amdgcn_wmma_f32_16x16x32_f16(
                          false, afr[kc], false, b0, (short)0, acc, false, false);
                b0 = b1; b1 = b2;
            }
            #pragma unroll
            for (int r = 0; r < 8; r++) {
                float v = acc[r];               // row r (lanes<16) / 8+r
                if (v > bestv[r]) { bestv[r] = v; besti[r] = col; }
            }
            roff = (roff == (NBUF - 1) * BBUF_BYTES) ? 0u : roff + BBUF_BYTES;
            woff = (woff == (NBUF - 1) * BBUF_BYTES) ? 0u : woff + BBUF_BYTES;
        }

        // ---- cross-lane argmax in each 16-lane group (tie -> low idx) ----
        #pragma unroll
        for (int r = 0; r < 8; r++) {
            #pragma unroll
            for (int m = 8; m >= 1; m >>= 1) {
                float ov = __shfl_xor(bestv[r], m, 32);
                int   oi = __shfl_xor(besti[r], m, 32);
                if (ov > bestv[r] || (ov == bestv[r] && oi < besti[r])) {
                    bestv[r] = ov; besti[r] = oi;
                }
            }
        }
        if (rowL == 0) {
            const int rbase = waveRow + ((lane < 16) ? 0 : 8);
            #pragma unroll
            for (int r = 0; r < 8; r++) {
                bestIdxLds[rbase + r] = besti[r];
                outIds[(size_t)(row0 + rbase + r) * QSTAGES + q] = (float)besti[r];
            }
        }
        __syncthreads();

        // ---- residual -= chosen (unnormalized, fp32) codeword ----
        float4* r4 = (float4*)res;
        for (int i = tid; i < TILE_M * DDIM / 4; i += THREADS) {
            const int row = i >> 7;             // DDIM/4 == 128
            const int d4  = i & 127;
            const float4 c = ((const float4*)(cb +
                ((size_t)q * KCODES + (size_t)bestIdxLds[row]) * DDIM))[d4];
            float4 v = r4[i];
            v.x -= c.x; v.y -= c.y; v.z -= c.z; v.w -= c.w;
            r4[i] = v;
        }
        __syncthreads();
    }

    // hard_x = x - res_final
    {
        const float4* xs = (const float4*)(x + (size_t)row0 * DDIM);
        float4* o4 = (float4*)(outHard + (size_t)row0 * DDIM);
        const float4* r4 = (const float4*)res;
        for (int i = tid; i < TILE_M * DDIM / 4; i += THREADS) {
            float4 a = xs[i], b = r4[i];
            float4 h; h.x = a.x - b.x; h.y = a.y - b.y;
                      h.z = a.z - b.z; h.w = a.w - b.w;
            o4[i] = h;
        }
    }
}

// ---------------------------------------------------------------------------
extern "C" void kernel_launch(void* const* d_in, const int* in_sizes, int n_in,
                              void* d_out, int out_size, void* d_ws, size_t ws_size,
                              hipStream_t stream) {
    (void)in_sizes; (void)n_in; (void)out_size; (void)ws_size;
    const float* x  = (const float*)d_in[0];          // [32768, 512] f32
    const float* cb = (const float*)d_in[1];          // [8, 4096, 512] f32
    _Float16* cb16  = (_Float16*)d_ws;                // 32 MB normalized f16

    float* outHard = (float*)d_out;                               // [N, D]
    float* outIds  = outHard + (size_t)N_ROWS * DDIM;             // [N, Q]

    // 1) normalize codebooks once (one wave per codeword, 8 waves/block)
    cb_norm_kernel<<<(QSTAGES * KCODES) / 8, 256, 0, stream>>>(cb, cb16);

    // 2) residual VQ: 256 blocks x 256 threads, ~313 KB dynamic LDS each
    const size_t shmem = (size_t)TILE_M * DDIM * sizeof(float)
                       + (size_t)NBUF * BBUF_BYTES;
    rvq_kernel<<<N_ROWS / TILE_M, THREADS, shmem, stream>>>(
        x, cb, cb16, outHard, outIds);
}
